// TransformerBlock_20907900797297
// MI455X (gfx1250) — compile-verified, matrix operand order checked by reading, not run on hardware
//
#include <hip/hip_runtime.h>
#include <hip/hip_bf16.h>

// ---------------------------------------------------------------------------
// TransformerBlock on gfx1250 (MI455X)
//   * all GEMMs: v_wmma_f32_16x16x32_bf16 (fp32 accumulate)
//   * A tiles staged via GLOBAL_LOAD_ASYNC_TO_LDS_B128 (ASYNCcnt)
//   * B tiles staged via TDM tensor_load_to_lds (TENSORcnt), LDS padding done
//     by the DMA engine (pad_interval=16dw, pad_amount=4dw -> 80B row pitch)
//   * double buffered; fragments consumed via ds_load_b128
// Shapes: B=4, S=2048, D=1024.  N_ROWS = B*S = 8192.
// ---------------------------------------------------------------------------

typedef __attribute__((ext_vector_type(16))) __bf16 v16bf;
typedef __attribute__((ext_vector_type(8)))  float  v8f;
typedef __attribute__((ext_vector_type(4)))  unsigned su32x4;
typedef __attribute__((ext_vector_type(8)))  unsigned su32x8;

#define B_SZ 4
#define S_SZ 2048
#define D_SZ 1024
#define NROWS 8192  // B*S

// LDS tile geometry: 128 rows x 32 bf16 cols, row pitch 80B (bank-spread, 16B aligned)
#define ROW_PITCH 80u
#define TILE_BYTES (128u * ROW_PITCH)       // 10240 B per tile
#define BUF_BYTES  (2u * TILE_BYTES)        // A tile + B tile per buffer
#define LDS_BYTES  (2u * BUF_BYTES)         // double buffered = 40960 B

union FragU { v16bf v; uint4 q[2]; };

__device__ __forceinline__ v8f wmma_bf16(v16bf a, v16bf b, v8f c) {
  return __builtin_amdgcn_wmma_f32_16x16x32_bf16(
      /*neg_a=*/false, a, /*neg_b=*/false, b,
      /*c_mod=*/(short)0, c, /*reuse_a=*/false, /*reuse_b=*/false);
}

// Async 16B copy global -> LDS (ASYNCcnt tracked).
__device__ __forceinline__ void async_ld16(unsigned lds_addr, const void* gptr) {
  asm volatile("global_load_async_to_lds_b128 %0, %1, off"
               :: "v"(lds_addr), "v"((unsigned long long)(uintptr_t)gptr)
               : "memory");
}
__device__ __forceinline__ void wait_async_le2() {
  asm volatile("s_wait_asynccnt 0x2" ::: "memory");
}
__device__ __forceinline__ void wait_async_0() {
  asm volatile("s_wait_asynccnt 0x0" ::: "memory");
}

// TDM: one descriptor moves a 128x32 bf16 tile (row stride = ld elements) from
// global to LDS, inserting 16B padding after every 64B row -> 80B LDS pitch.
// D# layout per ISA 8.3/8.4 (group0 = 4 SGPRs, group1 = 8 SGPRs).
__device__ __forceinline__ void tdm_load_tile(unsigned lds_addr, const void* gptr,
                                              unsigned ld_elems) {
  const unsigned long long ga = (unsigned long long)(uintptr_t)gptr;
  su32x4 g0;
  g0[0] = 1u;                                   // count=1 (valid user descriptor)
  g0[1] = lds_addr;                             // [63:32]  lds_addr (bytes)
  g0[2] = (unsigned)ga;                         // [95:64]  global_addr lo
  g0[3] = ((unsigned)(ga >> 32) & 0x01FFFFFFu)  // [120:96] global_addr hi (57-bit)
          | (2u << 30);                         // [127:126] type=2 ("image")
  su32x8 g1;
  g1[0] = (1u << 16)      // data_size = 2 bytes
        | (1u << 20)      // pad_enable
        | (3u << 22)      // pad_interval: 16 DWORDs (64B) between pads
        | (3u << 25);     // pad_amount:   4 DWORDs (16B)
  g1[1] = (ld_elems & 0xFFFFu) << 16;                      // tensor_dim0 lo16
  g1[2] = (ld_elems >> 16) | ((1u << 20) & 0xFFFFu) << 16; // tensor_dim0 hi / dim1 lo
  g1[2] = (ld_elems >> 16) | (0u << 16);                   // tensor_dim1 lo16 = 0 (of 1<<20)
  g1[3] = (1u << 4) | (32u << 16);   // tensor_dim1 hi16 = 16 (dim1 = 1<<20), tile_dim0 = 32
  g1[4] = 128u;                      // tile_dim1 = 128, tile_dim2 = 0
  g1[5] = ld_elems;                  // tensor_dim0_stride lo32 (elements)
  g1[6] = 0u;                        // stride hi16 = 0, tensor_dim1_stride lo16 = 0
  g1[7] = 0u;
  asm volatile("tensor_load_to_lds %0, %1" :: "s"(g0), "s"(g1) : "memory");
}

// Read one 16x32 bf16 fragment from an LDS tile (A-operand layout, ISA 7.12.2):
//   lane l (l<16):  row = r0 + l,      K chunks {0..7, 16..23}
//   lane l (l>=16): row = r0 + (l-16), K chunks {8..15, 24..31}
__device__ __forceinline__ v16bf lds_frag(const char* smem, unsigned tileBase,
                                          int r0, int lane) {
  const char* p = smem + tileBase + (unsigned)(r0 + (lane & 15)) * ROW_PITCH
                  + (unsigned)((lane >> 4) << 4);
  FragU u;
  u.q[0] = *reinterpret_cast<const uint4*>(p);        // K 0..7   (+8*hi)
  u.q[1] = *reinterpret_cast<const uint4*>(p + 32);   // K 16..23 (+8*hi)
  return u.v;
}

// ---------------------------------------------------------------------------
// Batched GEMM:  out[m,n] = sum_k A[m,k]*Bt[n,k]  (+bias) (+relu)
// Block = 256 threads = 8 waves; block tile 128(M) x 128(N); wave tile 32x64.
// K staged to LDS in 32-wide slabs: A via async-VMEM, B via TDM; double buffer.
// grid = (N/128, M/128, nbatch)
// ---------------------------------------------------------------------------
template <bool RELU, bool HAS_BIAS, bool OUT_F32, bool OUT_BF16, bool TRANS_BF16>
__global__ __launch_bounds__(256)
void gemm_bf16_kernel(const __hip_bfloat16* __restrict__ A, int lda, long long strideA,
                      const __hip_bfloat16* __restrict__ Bt, int ldb, long long strideB,
                      const float* __restrict__ bias,
                      float* __restrict__ outF, long long strideOF,
                      __hip_bfloat16* __restrict__ outB,
                      int M, int N, int K) {
  __shared__ char smem[LDS_BYTES];
  const unsigned ldsBase = (unsigned)(uintptr_t)&smem[0];  // LDS aperture: low 32 bits = offset

  const int z = blockIdx.z;
  A  += (size_t)z * strideA;
  Bt += (size_t)z * strideB;
  if (OUT_F32) outF += (size_t)z * strideOF;

  const int wave = threadIdx.x >> 5;
  const int lane = threadIdx.x & 31;
  const int t    = threadIdx.x;
  const int mBlock = blockIdx.y * 128;
  const int nBlock = blockIdx.x * 128;
  const int mSub = (wave & 3) * 32;   // wave's M offset within block tile
  const int nSub = (wave >> 2) * 64;  // wave's N offset within block tile

  // A staging map: 128x32 bf16 tile = 512 16B granules; thread t moves granules
  // t and t+256.  granule g: row = g>>2, 16B-chunk = g&3.
  const int r0 = t >> 2, c0 = t & 3;
  const int r1 = (t + 256) >> 2, c1 = (t + 256) & 3;

  auto stage = [&](int k, unsigned buf) {
    const unsigned aB = ldsBase + buf;
    async_ld16(aB + (unsigned)r0 * ROW_PITCH + (unsigned)c0 * 16u,
               A + (size_t)(mBlock + r0) * lda + k + c0 * 8);
    async_ld16(aB + (unsigned)r1 * ROW_PITCH + (unsigned)c1 * 16u,
               A + (size_t)(mBlock + r1) * lda + k + c1 * 8);
    if (wave == 0) {  // one TDM descriptor moves the whole 128x32 B tile
      tdm_load_tile(ldsBase + buf + TILE_BYTES,
                    Bt + (size_t)nBlock * ldb + k, (unsigned)ldb);
    }
  };

  v8f c[2][4] = {};

  stage(0, 0u);
  for (int k = 0; k < K; k += 32) {
    const unsigned p   = ((unsigned)k >> 5) & 1u;
    const unsigned cur = p * BUF_BYTES;
    if (k + 32 < K) {
      stage(k + 32, (p ^ 1u) * BUF_BYTES);            // fill other buffer
      wait_async_le2();                               // current A tile resident
      if (wave == 0) __builtin_amdgcn_s_wait_tensorcnt(1);  // current B tile resident
    } else {
      wait_async_0();
      if (wave == 0) __builtin_amdgcn_s_wait_tensorcnt(0);
    }
    if (k + 64 < K) {  // -> global_prefetch_b8, two slabs ahead
      __builtin_prefetch(A + (size_t)(mBlock + r0) * lda + k + 64, 0, 1);
    }
    __syncthreads();

    // Issue all 12 fragment reads first so WMMAs can run on partial dscnt waits.
    v16bf a0 = lds_frag(smem, cur, mSub, lane);
    v16bf a1 = lds_frag(smem, cur, mSub + 16, lane);
    v16bf b[4];
#pragma unroll
    for (int j = 0; j < 4; ++j)
      b[j] = lds_frag(smem, cur + TILE_BYTES, nSub + j * 16, lane);
#pragma unroll
    for (int j = 0; j < 4; ++j) {
      c[0][j] = wmma_bf16(a0, b[j], c[0][j]);
      c[1][j] = wmma_bf16(a1, b[j], c[1][j]);
    }
    __syncthreads();  // reads done before this buffer is re-staged
  }

  // C/D layout: VGPR r -> M = r + 8*(lane>>4);  N = lane&15 (per 16-wide tile)
  const int colL  = lane & 15;
  const int rowHi = (lane >> 4) << 3;
#pragma unroll
  for (int i = 0; i < 2; ++i) {
#pragma unroll
    for (int j = 0; j < 4; ++j) {
      const int n  = nBlock + nSub + j * 16 + colL;
      const float bv = HAS_BIAS ? bias[n] : 0.0f;
#pragma unroll
      for (int r = 0; r < 8; ++r) {
        const int m = mBlock + mSub + i * 16 + rowHi + r;
        float v = c[i][j][r] + bv;
        if (RELU) v = fmaxf(v, 0.0f);
        if (OUT_F32)  outF[(size_t)m * N + n] = v;
        if (OUT_BF16) outB[(size_t)m * N + n] = __float2bfloat16(v);
        if (TRANS_BF16) {
          // Vt[b][n][s]: b = m>>11, s = m&2047  (per-batch transpose store)
          const size_t idx = (((size_t)((m >> 11) * N + n)) << 11) + (size_t)(m & 2047);
          outB[idx] = __float2bfloat16(v);
        }
      }
    }
  }
}

// ---------------------------------------------------------------------------
// fp32 -> bf16 cast (grid-stride)
// ---------------------------------------------------------------------------
__global__ __launch_bounds__(256)
void cast_bf16_kernel(const float* __restrict__ in, __hip_bfloat16* __restrict__ out,
                      size_t n) {
  size_t i = (size_t)blockIdx.x * blockDim.x + threadIdx.x;
  const size_t stride = (size_t)gridDim.x * blockDim.x;
  for (; i < n; i += stride) out[i] = __float2bfloat16(in[i]);
}

// ---------------------------------------------------------------------------
// Transpose-cast: Wt[n][k] = bf16(W[k][n]),  W is Dm x Dm fp32. 32x32 LDS tile.
// block (32,8), grid (Dm/32, Dm/32)
// ---------------------------------------------------------------------------
__global__ __launch_bounds__(256)
void transpose_bf16_kernel(const float* __restrict__ W, __hip_bfloat16* __restrict__ Wt,
                           int Dm) {
  __shared__ float tile[32][33];
  const int x0 = blockIdx.x * 32, y0 = blockIdx.y * 32;
  const int tx = threadIdx.x, ty = threadIdx.y;
#pragma unroll
  for (int j = 0; j < 32; j += 8)
    tile[ty + j][tx] = W[(size_t)(y0 + ty + j) * Dm + (x0 + tx)];
  __syncthreads();
#pragma unroll
  for (int j = 0; j < 32; j += 8)
    Wt[(size_t)(x0 + ty + j) * Dm + (y0 + tx)] = __float2bfloat16(tile[tx][ty + j]);
}

// ---------------------------------------------------------------------------
// Causal softmax over rows of length S=2048 (in-place fp32 + bf16 copy).
// scale = 1/sqrt(D) = 1/32.  grid = B*S rows, 256 threads (8 cols/thread).
// ---------------------------------------------------------------------------
__global__ __launch_bounds__(256)
void softmax_causal_kernel(float* __restrict__ sc, __hip_bfloat16* __restrict__ wb) {
  const int S = S_SZ;
  const int row = blockIdx.x;
  const int q   = row & (S - 1);
  float* r = sc + (size_t)row * S;
  __hip_bfloat16* w = wb + (size_t)row * S;
  const int t = threadIdx.x;

  float vals[8];
  float mx = -3.0e38f;
#pragma unroll
  for (int i = 0; i < 8; ++i) {
    const int col = t + i * 256;
    const float v = (col <= q) ? r[col] * 0.03125f : -3.0e38f;
    vals[i] = v;
    mx = fmaxf(mx, v);
  }
  __shared__ float red[256];
  red[t] = mx; __syncthreads();
  for (int s2 = 128; s2 > 0; s2 >>= 1) {
    if (t < s2) red[t] = fmaxf(red[t], red[t + s2]);
    __syncthreads();
  }
  mx = red[0]; __syncthreads();

  float sum = 0.0f;
#pragma unroll
  for (int i = 0; i < 8; ++i) {
    const int col = t + i * 256;
    const float e = (col <= q) ? __expf(vals[i] - mx) : 0.0f;
    vals[i] = e;
    sum += e;
  }
  red[t] = sum; __syncthreads();
  for (int s2 = 128; s2 > 0; s2 >>= 1) {
    if (t < s2) red[t] += red[t + s2];
    __syncthreads();
  }
  const float inv = 1.0f / red[0];
#pragma unroll
  for (int i = 0; i < 8; ++i) {
    const int col = t + i * 256;
    const float o = vals[i] * inv;
    r[col] = o;
    w[col] = __float2bfloat16(o);
  }
}

// ---------------------------------------------------------------------------
// Fused residual add + LayerNorm over D=1024. grid = B*S rows, 256 threads.
// ---------------------------------------------------------------------------
template <bool OUT_BF16>
__global__ __launch_bounds__(256)
void add_layernorm_kernel(const float* __restrict__ X, const float* __restrict__ Y,
                          const float* __restrict__ g, const float* __restrict__ b,
                          float* __restrict__ outF, __hip_bfloat16* __restrict__ outB) {
  const int D = D_SZ;
  const size_t base = (size_t)blockIdx.x * D;
  const int t = threadIdx.x;

  float v[4];
  float s = 0.0f;
#pragma unroll
  for (int i = 0; i < 4; ++i) {
    const int col = t + i * 256;
    v[i] = X[base + col] + Y[base + col];
    s += v[i];
  }
  __shared__ float red[256];
  red[t] = s; __syncthreads();
  for (int s2 = 128; s2 > 0; s2 >>= 1) {
    if (t < s2) red[t] += red[t + s2];
    __syncthreads();
  }
  const float mu = red[0] * (1.0f / D);
  __syncthreads();

  float sv = 0.0f;
#pragma unroll
  for (int i = 0; i < 4; ++i) {
    const float d = v[i] - mu;
    sv += d * d;
  }
  red[t] = sv; __syncthreads();
  for (int s2 = 128; s2 > 0; s2 >>= 1) {
    if (t < s2) red[t] += red[t + s2];
    __syncthreads();
  }
  const float rs = rsqrtf(red[0] * (1.0f / D) + 1e-5f);
#pragma unroll
  for (int i = 0; i < 4; ++i) {
    const int col = t + i * 256;
    const float o = (v[i] - mu) * rs * g[col] + b[col];
    outF[base + col] = o;
    if (OUT_BF16) outB[base + col] = __float2bfloat16(o);
  }
}

// ---------------------------------------------------------------------------
// Host launch
// ---------------------------------------------------------------------------
extern "C" void kernel_launch(void* const* d_in, const int* in_sizes, int n_in,
                              void* d_out, int out_size, void* d_ws, size_t ws_size,
                              hipStream_t stream) {
  (void)in_sizes; (void)n_in; (void)out_size; (void)ws_size;

  const float* x  = (const float*)d_in[0];
  const float* Wq = (const float*)d_in[1];  const float* bq = (const float*)d_in[2];
  const float* Wk = (const float*)d_in[3];  const float* bk = (const float*)d_in[4];
  const float* Wv = (const float*)d_in[5];  const float* bv = (const float*)d_in[6];
  const float* W1 = (const float*)d_in[7];  const float* b1 = (const float*)d_in[8];
  const float* W2 = (const float*)d_in[9];  const float* b2 = (const float*)d_in[10];
  const float* g1 = (const float*)d_in[11]; const float* be1 = (const float*)d_in[12];
  const float* g2 = (const float*)d_in[13]; const float* be2 = (const float*)d_in[14];

  float* outX = (float*)d_out;                                   // [B,S,D]
  float* outW = (float*)d_out + (size_t)B_SZ * S_SZ * D_SZ;      // [B,S,S] (scores -> weights)

  // workspace carve-up (~202 MB)
  char* w = (char*)d_ws;
  auto take = [&](size_t bytes) { char* p = w; w += bytes; return p; };
  const size_t ACT_BF  = (size_t)NROWS * D_SZ * sizeof(__hip_bfloat16);  // 16 MB
  const size_t WT_BF   = (size_t)D_SZ * D_SZ * sizeof(__hip_bfloat16);   //  2 MB
  const size_t ATT_BF  = (size_t)NROWS * S_SZ * sizeof(__hip_bfloat16);  // 32 MB
  const size_t ACT_F32 = (size_t)NROWS * D_SZ * sizeof(float);           // 32 MB

  __hip_bfloat16* xb   = (__hip_bfloat16*)take(ACT_BF);
  __hip_bfloat16* WtQ  = (__hip_bfloat16*)take(WT_BF);
  __hip_bfloat16* WtK  = (__hip_bfloat16*)take(WT_BF);
  __hip_bfloat16* WtV  = (__hip_bfloat16*)take(WT_BF);
  __hip_bfloat16* Wt1  = (__hip_bfloat16*)take(WT_BF);
  __hip_bfloat16* Wt2  = (__hip_bfloat16*)take(WT_BF);
  __hip_bfloat16* Qb   = (__hip_bfloat16*)take(ACT_BF);
  __hip_bfloat16* Kb   = (__hip_bfloat16*)take(ACT_BF);
  __hip_bfloat16* Vt   = (__hip_bfloat16*)take(ACT_BF);   // [B, D, S] transposed
  __hip_bfloat16* wb   = (__hip_bfloat16*)take(ATT_BF);   // softmax weights bf16
  float*          attn = (float*)take(ACT_F32);           // attn_out; reused as ff2
  float*          h    = (float*)take(ACT_F32);
  __hip_bfloat16* hb   = (__hip_bfloat16*)take(ACT_BF);
  __hip_bfloat16* ff1b = (__hip_bfloat16*)take(ACT_BF);
  float*          ff2  = attn;  // alias: attn dead after LN1

  const dim3 blk(256);
  const dim3 tgrid(D_SZ / 32, D_SZ / 32);
  const dim3 tblk(32, 8);
  const dim3 gProj(D_SZ / 128, NROWS / 128, 1);    // 8192x1024 GEMMs
  const dim3 gScore(S_SZ / 128, S_SZ / 128, B_SZ); // per-batch 2048x2048
  const dim3 gAttn(D_SZ / 128, S_SZ / 128, B_SZ);  // per-batch 2048x1024

  // 0) casts / weight transposes
  cast_bf16_kernel<<<2048, blk, 0, stream>>>(x, xb, (size_t)NROWS * D_SZ);
  transpose_bf16_kernel<<<tgrid, tblk, 0, stream>>>(Wq, WtQ, D_SZ);
  transpose_bf16_kernel<<<tgrid, tblk, 0, stream>>>(Wk, WtK, D_SZ);
  transpose_bf16_kernel<<<tgrid, tblk, 0, stream>>>(Wv, WtV, D_SZ);
  transpose_bf16_kernel<<<tgrid, tblk, 0, stream>>>(W1, Wt1, D_SZ);
  transpose_bf16_kernel<<<tgrid, tblk, 0, stream>>>(W2, Wt2, D_SZ);

  // 1) QKV projections (bf16 WMMA, fp32 acc)
  gemm_bf16_kernel<false, true, false, true, false><<<gProj, blk, 0, stream>>>(
      xb, D_SZ, 0, WtQ, D_SZ, 0, bq, nullptr, 0, Qb, NROWS, D_SZ, D_SZ);
  gemm_bf16_kernel<false, true, false, true, false><<<gProj, blk, 0, stream>>>(
      xb, D_SZ, 0, WtK, D_SZ, 0, bk, nullptr, 0, Kb, NROWS, D_SZ, D_SZ);
  gemm_bf16_kernel<false, true, false, false, true><<<gProj, blk, 0, stream>>>(
      xb, D_SZ, 0, WtV, D_SZ, 0, bv, nullptr, 0, Vt, NROWS, D_SZ, D_SZ);

  // 2) scores = Q @ K^T (batched) -> raw scores into d_out weights slice
  gemm_bf16_kernel<false, false, true, false, false><<<gScore, blk, 0, stream>>>(
      Qb, D_SZ, (long long)S_SZ * D_SZ, Kb, D_SZ, (long long)S_SZ * D_SZ, nullptr,
      outW, (long long)S_SZ * S_SZ, nullptr, S_SZ, S_SZ, D_SZ);

  // 3) causal softmax (scale 1/32), in-place fp32 + bf16 copy
  softmax_causal_kernel<<<NROWS, blk, 0, stream>>>(outW, wb);

  // 4) attn = weights @ V (batched; Vt pre-transposed per batch)
  gemm_bf16_kernel<false, false, true, false, false><<<gAttn, blk, 0, stream>>>(
      wb, S_SZ, (long long)S_SZ * S_SZ, Vt, S_SZ, (long long)D_SZ * S_SZ, nullptr,
      attn, (long long)S_SZ * D_SZ, nullptr, S_SZ, D_SZ, S_SZ);

  // 5) h = LN(x + attn)
  add_layernorm_kernel<true><<<NROWS, blk, 0, stream>>>(x, attn, g1, be1, h, hb);

  // 6) ff1 = relu(h @ W1 + b1); ff2 = ff1 @ W2 + b2
  gemm_bf16_kernel<true, true, false, true, false><<<gProj, blk, 0, stream>>>(
      hb, D_SZ, 0, Wt1, D_SZ, 0, b1, nullptr, 0, ff1b, NROWS, D_SZ, D_SZ);
  gemm_bf16_kernel<false, true, true, false, false><<<gProj, blk, 0, stream>>>(
      ff1b, D_SZ, 0, Wt2, D_SZ, 0, b2, ff2, 0, nullptr, NROWS, D_SZ, D_SZ);

  // 7) out = LN(h + ff2)
  add_layernorm_kernel<false><<<NROWS, blk, 0, stream>>>(h, ff2, g2, be2, outX, nullptr);
}
